// SSIM_369367187797
// MI455X (gfx1250) — matrix operand order repaired and checked by reading, not compiled, not run on hardware
//
#include <hip/hip_runtime.h>
#include <hip/hip_bf16.h>
#include <math.h>

// ---------------------------------------------------------------------------
// SSIM (64x64 Gaussian window, separable) for MI455X / gfx1250.
// 1D convs are expressed as D = A(data 16x96) x T(Toeplitz gauss band 96x16)
// and run on v_wmma_f32_16x16x32_f16 (f16 operands, f32 accumulate).
// Horizontal pass writes its result transposed so the vertical pass reuses
// the identical row-contiguous A-load pattern.
// ---------------------------------------------------------------------------

#define WSZ    64
#define IMG_H  512
#define IMG_W  512
#define NBATCH 32
#define NOUT   449            // 512 - 64 + 1
#define NXT    29             // ceil(449/16)
#define OPAD   (NXT * 16)     // 464
#define NMAP   5
// sum_{d=-32}^{31} exp(-d^2 / (2*1.5^2)), precomputed (matches reference g.sum())
#define GNORM  3.75994129f

typedef _Float16 half_t;
typedef __attribute__((ext_vector_type(16))) _Float16 v16h;
typedef __attribute__((ext_vector_type(8)))  _Float16 v8h;
typedef __attribute__((ext_vector_type(8)))  float    v8f;

__device__ __forceinline__ float gauss_w(int d) {
    float t = (float)(d - 32);
    return __expf(-t * t / 4.5f) * (1.0f / GNORM);
}

// Fill LDS with the zero-extended gaussian band: gext[16+d] = g[d], d in [-16,95]
__device__ __forceinline__ void fill_gext(half_t* gext) {
    int t = threadIdx.x;
    if (t < 128) {
        int d = t - 16;
        float v = (d >= 0 && d < WSZ) ? gauss_w(d) : 0.0f;
        gext[t] = (half_t)v;
    }
    __syncthreads();
}

// Build the three K=32 Toeplitz B-chunks for this lane.
// B layout (wave32, f16 32x16): lanes 0-15 hold col n=lane, K in {0..7,16..23};
// lanes 16-31 hold col n=lane-16, K in {8..15,24..31} (matches A).
__device__ __forceinline__ void build_toeplitz(const half_t* gext, int lane,
                                               v16h& b0, v16h& b1, v16h& b2) {
    int n  = lane & 15;
    int kb = (lane >> 4) << 3;   // 0 or 8
#pragma unroll
    for (int j = 0; j < 16; ++j) {
        int idx = 16 + kb + j + ((j >= 8) ? 8 : 0) - n;           // in [1, 47]
        b0[j] = gext[idx];
        b1[j] = gext[idx + 32];
        b2[j] = gext[idx + 64];
    }
}

// A layout (wave32, f16 16x32): lanes 0-15 hold row M=lane, K in {0..7,16..23};
// lanes 16-31 hold row M=lane-16, K in {8..15,24..31}. Each run of 8 K-values
// is 8 contiguous halves in memory -> one 16B load.
__device__ __forceinline__ v16h load_a16(const half_t* rowp, int off0, int off1) {
    v8h pa = *reinterpret_cast<const v8h*>(rowp + off0);
    v8h pb = *reinterpret_cast<const v8h*>(rowp + off1);
    v16h a;
#pragma unroll
    for (int j = 0; j < 8; ++j) { a[j] = pa[j]; a[j + 8] = pb[j]; }
    return a;
}

// ---------------------------------------------------------------------------
__global__ void ssim_zero(float* facc, unsigned* uacc) {
    if (threadIdx.x == 0) {
        facc[0] = 0.0f;          // sum(contr)
        facc[1] = 0.0f;          // sum(si)
        uacc[2] = 0u;            // max bits (inputs are >= 0)
        uacc[3] = 0x7F800000u;   // min bits = +inf
    }
}

// f32 -> 5 f16 planes {a, b, a*a, b*b, a*b} + global max/min of img1
__global__ void __launch_bounds__(256) ssim_prep(const float* __restrict__ img1,
                                                 const float* __restrict__ img2,
                                                 half_t* __restrict__ P,
                                                 unsigned* __restrict__ uacc) {
    const size_t PL = (size_t)NBATCH * IMG_H * IMG_W;
    size_t i = (size_t)blockIdx.x * 256 + threadIdx.x;
    float a = img1[i];
    float c = img2[i];
    P[i]          = (half_t)a;
    P[PL + i]     = (half_t)c;
    P[2 * PL + i] = (half_t)(a * a);
    P[3 * PL + i] = (half_t)(c * c);
    P[4 * PL + i] = (half_t)(a * c);

    float mx = a, mn = a;
#pragma unroll
    for (int o = 16; o > 0; o >>= 1) {
        mx = fmaxf(mx, __shfl_xor(mx, o, 32));
        mn = fminf(mn, __shfl_xor(mn, o, 32));
    }
    if ((threadIdx.x & 31) == 0) {   // uint-bit order == float order for x >= 0
        atomicMax(&uacc[2], __float_as_uint(mx));
        atomicMin(&uacc[3], __float_as_uint(mn));
    }
}

// Horizontal 64-tap conv via WMMA; output stored TRANSPOSED: HT[m][b][x][y]
__global__ void __launch_bounds__(256) ssim_hconv(const half_t* __restrict__ P,
                                                  half_t* __restrict__ HT) {
    __shared__ half_t gext[128];
    fill_gext(gext);

    int lane = threadIdx.x & 31;
    int tile = blockIdx.x * 8 + (threadIdx.x >> 5);   // 148480 tiles total
    int xt = tile % NXT; tile /= NXT;
    int yt = tile & 31;  tile >>= 5;
    int b  = tile & 31;
    int m  = tile >> 5;
    int x0 = xt * 16, y0 = yt * 16;

    v16h b0, b1, b2;
    build_toeplitz(gext, lane, b0, b1, b2);

    int khalf = (lane >> 4) * 8;
    int row   = y0 + (lane & 15);
    const half_t* rowp =
        P + ((size_t)(m * NBATCH + b) * IMG_H + row) * IMG_W;

    v8f acc = {};
#pragma unroll
    for (int c = 0; c < 3; ++c) {
        int k0 = 32 * c + khalf;
        int xa = x0 + k0;      if (xa > IMG_W - 8) xa = IMG_W - 8;  // 8-aligned OOB runs carry g=0
        int xb = x0 + k0 + 16; if (xb > IMG_W - 8) xb = IMG_W - 8;
        v16h a = load_a16(rowp, xa, xb);
        const v16h bb = (c == 0) ? b0 : ((c == 1) ? b1 : b2);
        acc = __builtin_amdgcn_wmma_f32_16x16x32_f16(
                  false, a, false, bb, (short)0, acc, false, false);
    }

    // D[M=y-offset r+8*hi][N=x-offset lane%16] -> transposed store: 8 contiguous halves in y
    int hi   = lane >> 4;
    int xout = x0 + (lane & 15);
    half_t* hp = HT + (((size_t)(m * NBATCH + b) * OPAD + xout) * IMG_H + y0 + 8 * hi);
    v8h st;
#pragma unroll
    for (int r = 0; r < 8; ++r) st[r] = (half_t)acc[r];
    *reinterpret_cast<v8h*>(hp) = st;
}

// Vertical conv via WMMA for all 5 maps + fused SSIM elementwise + reduction
__global__ void __launch_bounds__(256) ssim_vconv(const half_t* __restrict__ HT,
                                                  float* __restrict__ facc,
                                                  const unsigned* __restrict__ uacc) {
    __shared__ half_t gext[128];
    fill_gext(gext);

    int lane = threadIdx.x & 31;
    int tile = blockIdx.x * 8 + (threadIdx.x >> 5);   // 26912 tiles total
    int xt = tile % NXT; tile /= NXT;
    int yt = tile % NXT;
    int b  = tile / NXT;
    int x0 = xt * 16, y0 = yt * 16;

    v16h b0, b1, b2;
    build_toeplitz(gext, lane, b0, b1, b2);

    int khalf = (lane >> 4) * 8;
    int xr    = x0 + (lane & 15);    // A row M = x-offset (HT row = fixed x, contiguous y)

    v8f accs[NMAP];
#pragma unroll
    for (int m = 0; m < NMAP; ++m) {
        const half_t* colp =
            HT + (((size_t)(m * NBATCH + b) * OPAD + xr) * IMG_H);
        v8f acc = {};
#pragma unroll
        for (int c = 0; c < 3; ++c) {
            int k0 = 32 * c + khalf;
            int ya = y0 + k0;      if (ya > IMG_H - 8) ya = IMG_H - 8;
            int yb = y0 + k0 + 16; if (yb > IMG_H - 8) yb = IMG_H - 8;
            v16h a = load_a16(colp, ya, yb);
            const v16h bb = (c == 0) ? b0 : ((c == 1) ? b1 : b2);
            acc = __builtin_amdgcn_wmma_f32_16x16x32_f16(
                      false, a, false, bb, (short)0, acc, false, false);
        }
        accs[m] = acc;
    }

    // dynamic range L from img1 stats (reference semantics)
    float mx = __uint_as_float(uacc[2]);
    float mn = __uint_as_float(uacc[3]);
    float Lv = (mx > 128.0f ? 255.0f : 1.0f) - (mn < -0.5f ? -1.0f : 0.0f);
    float C2 = (0.03f * Lv) * (0.03f * Lv);

    // D[M = x-offset = r+8*hi][N = y-offset = lane%16]
    int hi = lane >> 4;
    int y  = y0 + (lane & 15);
    float sc = 0.0f, ss = 0.0f;
    if (y < NOUT) {
#pragma unroll
        for (int r = 0; r < 8; ++r) {
            int x = x0 + r + 8 * hi;
            if (x < NOUT) {
                float mu1 = accs[0][r], mu2 = accs[1][r];
                float s1  = accs[2][r] - mu1 * mu1;
                float s2  = accs[3][r] - mu2 * mu2;
                float s12 = accs[4][r] - mu1 * mu2;
                float sd1 = sqrtf(s1), sd2 = sqrtf(s2);
                sc += (2.0f * sd1 * sd2 + C2) / (s1 + s2 + C2);
                ss += (s12 + C2) / (sd1 * sd2 + C2);
            }
        }
    }
#pragma unroll
    for (int o = 16; o > 0; o >>= 1) {
        sc += __shfl_xor(sc, o, 32);
        ss += __shfl_xor(ss, o, 32);
    }
    if (lane == 0) {
        atomicAdd(&facc[0], sc);
        atomicAdd(&facc[1], ss);
    }
}

__global__ void ssim_final(const float* __restrict__ facc, float* __restrict__ out) {
    if (threadIdx.x == 0) {
        float inv = 1.0f / ((float)NBATCH * (float)NOUT * (float)NOUT);
        out[0] = (facc[0] * inv) * sqrtf(facc[1] * inv);
    }
}

// ---------------------------------------------------------------------------
extern "C" void kernel_launch(void* const* d_in, const int* in_sizes, int n_in,
                              void* d_out, int out_size, void* d_ws, size_t ws_size,
                              hipStream_t stream) {
    (void)in_sizes; (void)n_in; (void)out_size; (void)ws_size;

    const float* img1 = (const float*)d_in[0];
    const float* img2 = (const float*)d_in[1];
    // d_in[2] (window) is regenerated on-device from the same closed form.
    float* out = (float*)d_out;

    // workspace layout:
    //   [0,256)                     : float/uint accumulators
    //   P : 5 f16 planes            : 5*32*512*512 halves (~84 MB)
    //   HT: 5 transposed H planes   : 5*32*464*512 halves (~76 MB)
    char*     wsb  = (char*)d_ws;
    float*    facc = (float*)wsb;
    unsigned* uacc = (unsigned*)wsb;
    half_t*   P    = (half_t*)(wsb + 256);
    half_t*   HT   = P + (size_t)NMAP * NBATCH * IMG_H * IMG_W;

    ssim_zero<<<1, 64, 0, stream>>>(facc, uacc);

    const int nElem  = NBATCH * IMG_H * IMG_W;               // 8,388,608
    ssim_prep<<<nElem / 256, 256, 0, stream>>>(img1, img2, P, uacc);

    const int hTiles = NMAP * NBATCH * (IMG_H / 16) * NXT;   // 148,480
    ssim_hconv<<<hTiles / 8, 256, 0, stream>>>(P, HT);

    const int vTiles = NBATCH * NXT * NXT;                   // 26,912
    ssim_vconv<<<vTiles / 8, 256, 0, stream>>>(HT, facc, uacc);

    ssim_final<<<1, 32, 0, stream>>>(facc, out);
}
